// MultiheadAttentionTransparent_23141283791578
// MI455X (gfx1250) — matrix operand order repaired and checked by reading, not compiled, not run on hardware
//
#include <hip/hip_runtime.h>
#include <hip/hip_bf16.h>
#include <cstdint>

// ---------------------------------------------------------------------------
// MHA forward for MI455X (gfx1250, wave32, WMMA f32_16x16x32_f16 + TDM)
// B=4, T=S=2048, E=1024, H=16, D=64
// ---------------------------------------------------------------------------

#define BATCH 4
#define SEQ   2048
#define EMB   1024
#define HEADS 16
#define HDIM  64

typedef __attribute__((ext_vector_type(16))) _Float16 v16h;
typedef __attribute__((ext_vector_type(8)))  _Float16 v8h;
typedef __attribute__((ext_vector_type(4)))  _Float16 v4h;
typedef __attribute__((ext_vector_type(8)))  float    v8f;
typedef unsigned int u32x4 __attribute__((ext_vector_type(4)));
typedef int          i32x8 __attribute__((ext_vector_type(8)));
typedef int          i32x4 __attribute__((ext_vector_type(4)));

static __device__ inline v16h cat16(v8h lo, v8h hi) {
  v16h r;
#pragma unroll
  for (int i = 0; i < 8; ++i) { r[i] = lo[i]; r[i + 8] = hi[i]; }
  return r;
}

// A-operand fragment (16x32 f16): lane holds row M=lane%16,
// K halves = (lane/16)*8 + {0..7} and 16 + (lane/16)*8 + {0..7}.
static __device__ inline v16h load_a_frag(const _Float16* base) {
  return cat16(*reinterpret_cast<const v8h*>(base),
               *reinterpret_cast<const v8h*>(base + 16));
}

// B-operand fragment (32x16 f16, NT source B[N][K] row-major): lane holds
// col N=lane%16, K halves = (lane/16)*8 + {0..7} and 16 + (lane/16)*8 + {0..7}.
static __device__ inline v16h load_b_frag(const _Float16* base) {
  return cat16(*reinterpret_cast<const v8h*>(base),
               *reinterpret_cast<const v8h*>(base + 16));
}

static __device__ inline v8f wmma_f16(v16h a, v16h b, v8f c) {
  return __builtin_amdgcn_wmma_f32_16x16x32_f16(false, a, false, b,
                                                (short)0, c, false, false);
}

// ---------------------------------------------------------------------------
// Tensor Data Mover: async 2D f16 tile (tile_rows x tile_k) from row-major
// global tensor (row stride = row_stride elems) into contiguous LDS.
// D# per CDNA5 ISA ch.8: group0 = {count/lds/global/type}, group1 = dims.
// ---------------------------------------------------------------------------
static __device__ inline void tdm_load_tile_f16(unsigned lds_addr,
                                                const void* gaddr,
                                                unsigned tile_k,
                                                unsigned tile_rows,
                                                unsigned tensor_dim0,
                                                unsigned row_stride) {
  unsigned long long ga = (unsigned long long)(uintptr_t)gaddr;
  u32x4 g0;
  g0[0] = 1u;                                   // count=1 (valid descriptor)
  g0[1] = lds_addr;                             // lds_addr  [63:32]
  g0[2] = (unsigned)(ga & 0xffffffffu);         // global_addr[95:64]
  g0[3] = (unsigned)((ga >> 32) & 0x01ffffffu)  // global_addr[120:96]
          | (2u << 30);                         // type=2 ("image") [127:126]
  i32x8 g1;
  g1[0] = (int)(1u << 16);                      // data_size=1 -> 2 bytes
  g1[1] = (int)((tensor_dim0 & 0xffffu) << 16); // tensor_dim0[15:0] @ bit48
  g1[2] = (int)((tensor_dim0 >> 16) |           // tensor_dim0[31:16]
                ((tile_rows & 0xffffu) << 16)); // tensor_dim1[15:0] @ bit80
  g1[3] = (int)((tile_k & 0xffffu) << 16);      // tile_dim0 @ bit112
  g1[4] = (int)(tile_rows & 0xffffu);           // tile_dim1 @ bit128
  g1[5] = (int)row_stride;                      // tensor_dim0_stride[31:0]
  g1[6] = 0;                                    // stride hi + dim1_stride lo
  g1[7] = 0;
  i32x4 z4 = (i32x4){0, 0, 0, 0};
#if __clang_major__ >= 23
  i32x8 z8 = (i32x8){0, 0, 0, 0, 0, 0, 0, 0};
  __builtin_amdgcn_tensor_load_to_lds(g0, g1, z4, z4, z8, 0);
#else
  __builtin_amdgcn_tensor_load_to_lds(g0, g1, z4, z4, 0);
#endif
}

// ---------------------------------------------------------------------------
// Elementwise helpers
// ---------------------------------------------------------------------------
__global__ void cvt_f32_to_f16_kernel(const float* __restrict__ src,
                                      _Float16* __restrict__ dst, size_t n) {
  size_t i = (size_t)blockIdx.x * blockDim.x + threadIdx.x;
  size_t stride = (size_t)gridDim.x * blockDim.x;
  for (; i < n; i += stride) dst[i] = (_Float16)src[i];
}

__global__ void zero_f32_kernel(float* __restrict__ p, size_t n) {
  size_t i = (size_t)blockIdx.x * blockDim.x + threadIdx.x;
  size_t stride = (size_t)gridDim.x * blockDim.x;
  for (; i < n; i += stride) p[i] = 0.0f;
}

// vh [B,S,H,D] f16 -> vt [B,H,D,S] f16 so PV GEMM is NT (K=S contiguous)
__global__ void transpose_v_kernel(const _Float16* __restrict__ vh,
                                   _Float16* __restrict__ vt) {
  size_t i = (size_t)blockIdx.x * blockDim.x + threadIdx.x;
  const size_t n = (size_t)BATCH * SEQ * EMB;
  if (i >= n) return;
  int d = (int)(i & (HDIM - 1));
  int h = (int)((i >> 6) & (HEADS - 1));
  int s = (int)((i >> 10) & (SEQ - 1));
  int b = (int)(i >> 21);
  vt[(((size_t)b * HEADS + h) * HDIM + d) * SEQ + s] = vh[i];
}

// ---------------------------------------------------------------------------
// Blocked NT GEMM: C[M,N] = scale * (A[M,K] @ Bw[N,K]^T) (+ bias)
// BM=BN=128, BK=32. 256 threads = 8 waves (4 along M x 2 along N),
// each wave computes a 32x64 sub-tile via 2x4 WMMA fragments.
// B tile (and A tile when f16) staged through the Tensor Data Mover.
// ---------------------------------------------------------------------------
template <bool A_IS_F32, bool OUT_F32, bool HAS_BIAS>
__global__ void __launch_bounds__(256)
gemm_nt_kernel(const void* __restrict__ Aptr,
               const _Float16* __restrict__ Bw,
               void* __restrict__ Cptr,
               const float* __restrict__ bias,
               int M, int N, int K, float scale) {
  __shared__ _Float16 As[128 * 32];
  __shared__ _Float16 Bs[128 * 32];

  const int tid  = threadIdx.x;
  const int wave = tid >> 5;
  const int lane = tid & 31;
  const int lrow = lane & 15;
  const int lhi  = lane >> 4;

  const int bm = blockIdx.x * 128;
  const int bn = blockIdx.y * 128;
  const int wm = (wave & 3) * 32;
  const int wn = (wave >> 2) * 64;

  const unsigned ldsA = (unsigned)(uintptr_t)(void*)As;
  const unsigned ldsB = (unsigned)(uintptr_t)(void*)Bs;

  v8f acc[2][4];
#pragma unroll
  for (int i = 0; i < 2; ++i)
#pragma unroll
    for (int j = 0; j < 4; ++j) acc[i][j] = (v8f){};

  for (int k0 = 0; k0 < K; k0 += 32) {
    // ---- stage A tile [128 x 32]
    if (A_IS_F32) {
      const float* A = (const float*)Aptr;
#pragma unroll
      for (int it = 0; it < 4; ++it) {
        int r  = it * 32 + (tid >> 3);
        int c4 = (tid & 7) * 4;
        float4 v = *reinterpret_cast<const float4*>(
            A + (size_t)(bm + r) * K + k0 + c4);
        v4h hv;
        hv[0] = (_Float16)v.x; hv[1] = (_Float16)v.y;
        hv[2] = (_Float16)v.z; hv[3] = (_Float16)v.w;
        *reinterpret_cast<v4h*>(&As[r * 32 + c4]) = hv;   // one b64 LDS store
      }
    }
    // ---- TDM stages: B tile always; A tile too when already f16
    if (wave == 0) {
      tdm_load_tile_f16(ldsB, Bw + (size_t)bn * K + k0, 32, 128, K, K);
      if (!A_IS_F32) {
        tdm_load_tile_f16(ldsA,
                          (const _Float16*)Aptr + (size_t)bm * K + k0,
                          32, 128, K, K);
      }
    }
    __builtin_amdgcn_s_wait_tensorcnt(0);
    __syncthreads();

    v16h af[2], bf[4];
#pragma unroll
    for (int i = 0; i < 2; ++i)
      af[i] = load_a_frag(&As[(wm + i * 16 + lrow) * 32 + lhi * 8]);
#pragma unroll
    for (int j = 0; j < 4; ++j)
      bf[j] = load_b_frag(&Bs[(wn + j * 16 + lrow) * 32 + lhi * 8]);

#pragma unroll
    for (int i = 0; i < 2; ++i)
#pragma unroll
      for (int j = 0; j < 4; ++j)
        acc[i][j] = wmma_f16(af[i], bf[j], acc[i][j]);

    __syncthreads();
  }

  // ---- epilogue: C-frag row = (lane/16)*8 + r, col = lane%16
#pragma unroll
  for (int i = 0; i < 2; ++i) {
#pragma unroll
    for (int j = 0; j < 4; ++j) {
      int col = bn + wn + j * 16 + lrow;
      float bv = HAS_BIAS ? bias[col] : 0.0f;
#pragma unroll
      for (int r = 0; r < 8; ++r) {
        int row = bm + wm + i * 16 + lhi * 8 + r;
        float v = acc[i][j][r] * scale + bv;
        if (OUT_F32)
          ((float*)Cptr)[(size_t)row * N + col] = v;
        else
          ((_Float16*)Cptr)[(size_t)row * N + col] = (_Float16)v;
      }
    }
  }
}

// ---------------------------------------------------------------------------
// Fused attention per (b, h, 16-row tile of T):
//   scores = q16 @ K^T (WMMA) -> LDS f16 [16][2048]
//   softmax in place (f32 math, b128 LDS sweeps), atomic-accumulate avg/H
//   out16  = P @ V (WMMA on vt [D][S]), cross-wave LDS reduction
// ---------------------------------------------------------------------------
__global__ void __launch_bounds__(256)
attn_kernel(const _Float16* __restrict__ qh,
            const _Float16* __restrict__ kh,
            const _Float16* __restrict__ vt,
            _Float16* __restrict__ out_heads,
            float* __restrict__ avg_out) {
  __shared__ _Float16 sP[16 * SEQ];         // scores then probs (64 KB)
  __shared__ float    sRed[8 * 16 * HDIM];  // PV partials (32 KB)

  const int tid  = threadIdx.x;
  const int wave = tid >> 5;
  const int lane = tid & 31;
  const int lrow = lane & 15;
  const int lhi  = lane >> 4;

  const int blk = blockIdx.x;
  const int b   = blk >> 11;
  const int h   = (blk >> 7) & (HEADS - 1);
  const int t0  = (blk & 127) << 4;

  // ---- Q fragments for this 16-row tile (K-dim = D = 64 -> 2 frags)
  const _Float16* qbase =
      qh + ((size_t)(b * SEQ + t0 + lrow)) * EMB + h * HDIM;
  v16h qf0 = load_a_frag(qbase + lhi * 8);
  v16h qf1 = load_a_frag(qbase + 32 + lhi * 8);

  // ---- Phase 1: scores -> sP (waves split S into 16-col tiles)
  for (int st = wave; st < SEQ / 16; st += 8) {
    int s0 = st * 16;
    const _Float16* kbase =
        kh + ((size_t)(b * SEQ + s0 + lrow)) * EMB + h * HDIM;
    if (st + 8 < SEQ / 16)
      __builtin_prefetch(kbase + (size_t)128 * EMB, 0, 1);
    v16h kf0 = load_b_frag(kbase + lhi * 8);
    v16h kf1 = load_b_frag(kbase + 32 + lhi * 8);
    v8f acc = (v8f){};
    acc = wmma_f16(qf0, kf0, acc);
    acc = wmma_f16(qf1, kf1, acc);
#pragma unroll
    for (int r = 0; r < 8; ++r)
      sP[(lhi * 8 + r) * SEQ + s0 + lrow] = (_Float16)acc[r];
  }
  __syncthreads();

  // ---- Phase 2: softmax per row (wave handles 2 rows), in place in sP.
  // Lane owns interleaved 16-byte chunks -> conflict-free ds_load_b128.
  for (int rr = 0; rr < 2; ++rr) {
    int row = wave * 2 + rr;
    _Float16* prow = &sP[row * SEQ];

    float mx = -3.0e38f;
#pragma unroll
    for (int i = 0; i < 8; ++i) {
      v8h ch = *reinterpret_cast<const v8h*>(prow + (lane + 32 * i) * 8);
#pragma unroll
      for (int e = 0; e < 8; ++e) mx = fmaxf(mx, (float)ch[e]);
    }
#pragma unroll
    for (int off = 16; off > 0; off >>= 1)
      mx = fmaxf(mx, __shfl_xor(mx, off, 32));

    float sum = 0.0f;
#pragma unroll
    for (int i = 0; i < 8; ++i) {
      int sb = (lane + 32 * i) * 8;
      v8h ch = *reinterpret_cast<const v8h*>(prow + sb);
      v8h eo;
#pragma unroll
      for (int e = 0; e < 8; ++e) {
        float ev = __expf((float)ch[e] - mx);
        sum += ev;
        eo[e] = (_Float16)ev;
      }
      *reinterpret_cast<v8h*>(prow + sb) = eo;
    }
#pragma unroll
    for (int off = 16; off > 0; off >>= 1)
      sum += __shfl_xor(sum, off, 32);
    float inv = 1.0f / sum;

    float* avg = avg_out + ((size_t)b * SEQ + t0 + row) * SEQ;
#pragma unroll
    for (int i = 0; i < 8; ++i) {
      int sb = (lane + 32 * i) * 8;
      v8h ch = *reinterpret_cast<const v8h*>(prow + sb);
      v8h po;
#pragma unroll
      for (int e = 0; e < 8; ++e) {
        float p = (float)ch[e] * inv;
        po[e] = (_Float16)p;
        atomicAdd(&avg[sb + e], p * (1.0f / (float)HEADS));
      }
      *reinterpret_cast<v8h*>(prow + sb) = po;
    }
  }
  __syncthreads();

  // ---- Phase 3: out16 = P @ V, waves split K=S range (256 each)
  v8f oacc[4];
#pragma unroll
  for (int j = 0; j < 4; ++j) oacc[j] = (v8f){};

  const _Float16* vthead = vt + ((size_t)(b * HEADS + h)) * HDIM * SEQ;
  for (int ks = 0; ks < 8; ++ks) {
    int s0 = wave * 256 + ks * 32;
    v16h pf = load_a_frag(&sP[lrow * SEQ + s0 + lhi * 8]);
#pragma unroll
    for (int j = 0; j < 4; ++j) {
      const _Float16* vbase =
          vthead + (size_t)(j * 16 + lrow) * SEQ + s0 + lhi * 8;
      oacc[j] = wmma_f16(pf, load_b_frag(vbase), oacc[j]);
    }
  }
#pragma unroll
  for (int j = 0; j < 4; ++j)
#pragma unroll
    for (int r = 0; r < 8; ++r)
      sRed[wave * (16 * HDIM) + (lhi * 8 + r) * HDIM + j * 16 + lrow] =
          oacc[j][r];
  __syncthreads();

  for (int e = tid; e < 16 * HDIM; e += 256) {
    float s = 0.0f;
#pragma unroll
    for (int w = 0; w < 8; ++w) s += sRed[w * (16 * HDIM) + e];
    int row = e >> 6, col = e & (HDIM - 1);
    out_heads[((size_t)(b * SEQ + t0 + row)) * EMB + h * HDIM + col] =
        (_Float16)s;
  }
}

// ---------------------------------------------------------------------------
// Host-side orchestration
// ---------------------------------------------------------------------------
extern "C" void kernel_launch(void* const* d_in, const int* in_sizes, int n_in,
                              void* d_out, int out_size, void* d_ws,
                              size_t ws_size, hipStream_t stream) {
  (void)in_sizes; (void)n_in; (void)out_size; (void)ws_size;

  const float* query = (const float*)d_in[0];
  const float* key_  = (const float*)d_in[1];
  const float* value = (const float*)d_in[2];
  const float* Wq = (const float*)d_in[3];
  const float* Wk = (const float*)d_in[4];
  const float* Wv = (const float*)d_in[5];
  const float* Wo = (const float*)d_in[6];
  const float* bo = (const float*)d_in[7];

  const size_t MT  = (size_t)BATCH * SEQ;
  const size_t ACT = MT * EMB;
  const size_t WEL = (size_t)EMB * EMB;

  char* ws = (char*)d_ws;
  _Float16* Wq_h = (_Float16*)(ws);
  _Float16* Wk_h = Wq_h + WEL;
  _Float16* Wv_h = Wk_h + WEL;
  _Float16* Wo_h = Wv_h + WEL;
  _Float16* qh   = Wo_h + WEL;
  _Float16* kh   = qh + ACT;
  _Float16* vh   = kh + ACT;
  _Float16* vt   = vh + ACT;
  _Float16* oh   = vt + ACT;

  float* out_total = (float*)d_out;       // [B,T,E]
  float* out_avg   = out_total + ACT;     // [B,T,S]

  // 1) convert weights to f16
  cvt_f32_to_f16_kernel<<<4096, 256, 0, stream>>>(Wq, Wq_h, WEL);
  cvt_f32_to_f16_kernel<<<4096, 256, 0, stream>>>(Wk, Wk_h, WEL);
  cvt_f32_to_f16_kernel<<<4096, 256, 0, stream>>>(Wv, Wv_h, WEL);
  cvt_f32_to_f16_kernel<<<4096, 256, 0, stream>>>(Wo, Wo_h, WEL);

  // 2) projections (NT GEMMs, f32 A converted on the fly, f16 out)
  dim3 ggrid(MT / 128, EMB / 128);
  const float scaling = 0.125f;  // D^-0.5, D=64
  gemm_nt_kernel<true, false, false><<<ggrid, 256, 0, stream>>>(
      query, Wq_h, qh, nullptr, (int)MT, EMB, EMB, scaling);
  gemm_nt_kernel<true, false, false><<<ggrid, 256, 0, stream>>>(
      key_, Wk_h, kh, nullptr, (int)MT, EMB, EMB, 1.0f);
  gemm_nt_kernel<true, false, false><<<ggrid, 256, 0, stream>>>(
      value, Wv_h, vh, nullptr, (int)MT, EMB, EMB, 1.0f);

  // 3) transpose V per head -> [B,H,D,S]
  transpose_v_kernel<<<(unsigned)(ACT / 256), 256, 0, stream>>>(vh, vt);

  // 4) zero avg-weights accumulator, then fused attention
  const size_t AVGN = (size_t)BATCH * SEQ * SEQ;
  zero_f32_kernel<<<8192, 256, 0, stream>>>(out_avg, AVGN);
  attn_kernel<<<BATCH * HEADS * (SEQ / 16), 256, 0, stream>>>(
      qh, kh, vt, oh, out_avg);

  // 5) output projection (f16 A, f32 out, +bias)
  gemm_nt_kernel<false, true, true><<<ggrid, 256, 0, stream>>>(
      oh, Wo_h, out_total, bo, (int)MT, EMB, EMB, 1.0f);
}
